// BGraphConvWrapper_75265006895802
// MI455X (gfx1250) — compile-verified, matrix operand order checked by reading, not run on hardware
//
#include <hip/hip_runtime.h>
#include <math.h>

#define DIM      128
#define NNODES   32000
#define NEDGES   256000
#define NLAYERS  2
#define LN_EPS   1e-5f

typedef __attribute__((ext_vector_type(16))) __bf16 v16bf;
typedef __attribute__((ext_vector_type(8)))  float  v8f;

__device__ __forceinline__ float gelu_exact(float x) {
    return 0.5f * x * (1.0f + erff(x * 0.70710678118654752f));
}

// K position inside a 32-wide chunk for the CDNA5 16-bit WMMA A/B fragment.
// VGPR i (pair 2i,2i+1): i<4 -> K = 8*half + 2i ; i>=4 -> K = 16 + 8*half + 2(i-4)
__device__ __forceinline__ int frag_k(int i, int half) {
    return (i < 4) ? (8 * half + 2 * i) : (16 + 8 * half + 2 * (i - 4));
}

// Load one 16(row) x 32(K) bf16 fragment from a row-major f32 matrix (cvt on load).
__device__ __forceinline__ v16bf load_frag(const float* __restrict__ M, int ld,
                                           int rowBase, int kBase, int lane) {
    const int half = lane >> 4;
    const float* rp = M + (size_t)(rowBase + (lane & 15)) * ld + kBase;
    v16bf r;
#pragma unroll
    for (int i = 0; i < 8; ++i) {
        int k = frag_k(i, half);
        r[2 * i]     = (__bf16)rp[k];
        r[2 * i + 1] = (__bf16)rp[k + 1];
    }
    return r;
}

// Same, but source already bf16 (global or LDS pointer).
__device__ __forceinline__ v16bf load_frag_bf16(const __bf16* M, int ld,
                                                int rowBase, int kBase, int lane) {
    const int half = lane >> 4;
    const __bf16* rp = M + (size_t)(rowBase + (lane & 15)) * ld + kBase;
    v16bf r;
#pragma unroll
    for (int i = 0; i < 8; ++i) {
        int k = frag_k(i, half);
        r[2 * i]     = rp[k];
        r[2 * i + 1] = rp[k + 1];
    }
    return r;
}

// ---------------------------------------------------------------------------
// Weight folding:  W2' = Wm2 * We (stored bf16),  W3' = Wm3 * Wn (f32),
//                  b'  = bm + Wm2*be + Wm3*bn
// Wm is [128, 384] row-major (cols 0..127 = Wm1, 128..255 = Wm2, 256..383 = Wm3)
// ---------------------------------------------------------------------------
__global__ __launch_bounds__(128) void fold_kernel(
    const float* __restrict__ Wm, const float* __restrict__ bm,
    const float* __restrict__ We, const float* __restrict__ be,
    const float* __restrict__ Wn, const float* __restrict__ bn,
    __bf16* __restrict__ W2bf, float* __restrict__ W3, float* __restrict__ bp) {
    int o = blockIdx.x;   // output row   0..127
    int k = threadIdx.x;  // output col   0..127
    const float* wm2 = Wm + (size_t)o * (3 * DIM) + DIM;
    const float* wm3 = Wm + (size_t)o * (3 * DIM) + 2 * DIM;
    float a2 = 0.0f, a3 = 0.0f;
    for (int j = 0; j < DIM; ++j) {
        a2 = fmaf(wm2[j], We[(size_t)j * DIM + k], a2);
        a3 = fmaf(wm3[j], Wn[(size_t)j * DIM + k], a3);
    }
    W2bf[o * DIM + k] = (__bf16)a2;
    W3[o * DIM + k] = a3;
    if (k == 0) {
        float bv = bm[o];
        for (int j = 0; j < DIM; ++j) bv = fmaf(wm2[j], be[j], fmaf(wm3[j], bn[j], bv));
        bp[o] = bv;
    }
}

__global__ void zero_kernel(float* __restrict__ p, int n) {
    int i = blockIdx.x * blockDim.x + threadIdx.x;
    if (i < n) p[i] = 0.0f;
}

__global__ void cvt_bf16_kernel(const float* __restrict__ s, __bf16* __restrict__ d, int n) {
    int i = blockIdx.x * blockDim.x + threadIdx.x;
    if (i < n) d[i] = (__bf16)s[i];
}

// Degree counts for scatter-mean (depend only on edge_index -> once per call)
__global__ void count_kernel(const int* __restrict__ ei,
                             float* __restrict__ cnt_fw, float* __restrict__ cnt_bw) {
    int e = blockIdx.x * blockDim.x + threadIdx.x;
    if (e < NEDGES) {
        atomicAdd(&cnt_fw[ei[NEDGES + e]], 1.0f);  // fw aggregates at dst
        atomicAdd(&cnt_bw[ei[e]], 1.0f);           // bw aggregates at src
    }
}

// ---------------------------------------------------------------------------
// Per-node contributions: A_* = x @ Wm1^T  (dst term),  C_* = x @ W3'^T (src term)
// One wave per 16-node x 16-out tile.
// ---------------------------------------------------------------------------
__global__ __launch_bounds__(256) void node_gemm_kernel(
    const float* __restrict__ x,
    const float* __restrict__ Wm_fw, const float* __restrict__ W3_fw,
    const float* __restrict__ Wm_bw, const float* __restrict__ W3_bw,
    float* __restrict__ A_fw, float* __restrict__ C_fw,
    float* __restrict__ A_bw, float* __restrict__ C_bw) {
    int wave = (blockIdx.x << 3) | (threadIdx.x >> 5);
    int lane = threadIdx.x & 31;
    int mat = wave & 3;
    int colTile = (wave >> 2) & 7;
    int nodeTile = wave >> 5;
    if (nodeTile >= NNODES / 16) return;

    const float* W;
    int ldw;
    float* out;
    switch (mat) {
        case 0:  W = Wm_fw; ldw = 3 * DIM; out = A_fw; break;  // Wm1_fw (first 128 cols)
        case 1:  W = W3_fw; ldw = DIM;     out = C_fw; break;
        case 2:  W = Wm_bw; ldw = 3 * DIM; out = A_bw; break;
        default: W = W3_bw; ldw = DIM;     out = C_bw; break;
    }
    int rowBase = nodeTile * 16, colBase = colTile * 16;
    v8f acc = {};
#pragma unroll
    for (int c = 0; c < 4; ++c) {
        v16bf a = load_frag(x, DIM, rowBase, c * 32, lane);
        v16bf b = load_frag(W, ldw, colBase, c * 32, lane);
        acc = __builtin_amdgcn_wmma_f32_16x16x32_bf16(false, a, false, b,
                                                      (short)0, acc, false, false);
    }
    int n = colBase + (lane & 15);
    int half = lane >> 4;
#pragma unroll
    for (int r = 0; r < 8; ++r)
        out[(size_t)(rowBase + r + 8 * half) * DIM + n] = acc[r];
}

// ---------------------------------------------------------------------------
// Edge messages. One wave handles one 16-edge tile x ALL 128 outputs for one
// direction (8 accumulators, 32 WMMAs). Weights (bf16) + bias staged in LDS
// once per block; edge_attr read once (bf16) per K-chunk per wave.
// Grid: 2000 blocks/dir * 2 dirs; 8 waves/block, wave w handles eTile
// (blockIdx>>1)*8 + w.
// ---------------------------------------------------------------------------
__global__ __launch_bounds__(256) void edge_msg_kernel(
    const __bf16* __restrict__ ea16, const int* __restrict__ ei,
    const __bf16* __restrict__ W2b_fw, const __bf16* __restrict__ W2b_bw,
    const float* __restrict__ b_fw,  const float* __restrict__ b_bw,
    const float* __restrict__ A_fw,  const float* __restrict__ C_fw,
    const float* __restrict__ A_bw,  const float* __restrict__ C_bw,
    float* __restrict__ sum_fw, float* __restrict__ sum_bw) {
    __shared__ __bf16 sW[DIM * DIM];  // 32 KB of the 320 KB WGP LDS
    __shared__ float  sB[DIM];

    const int dir  = blockIdx.x & 1;
    const int lane = threadIdx.x & 31;
    const int waveId = threadIdx.x >> 5;

    const __bf16* W2 = dir ? W2b_bw : W2b_fw;
    const float*  bb = dir ? b_bw : b_fw;
    const float*  A  = dir ? A_bw : A_fw;
    const float*  C  = dir ? C_bw : C_fw;
    float*        S  = dir ? sum_bw : sum_fw;

    {   // stage folded weight matrix + bias into LDS
        const uint32_t* src = reinterpret_cast<const uint32_t*>(W2);
        uint32_t* dst = reinterpret_cast<uint32_t*>(sW);
        for (int i = threadIdx.x; i < DIM * DIM / 2; i += 256) dst[i] = src[i];
        if (threadIdx.x < DIM) sB[threadIdx.x] = bb[threadIdx.x];
    }
    __syncthreads();

    const int eTile = (blockIdx.x >> 1) * 8 + waveId;   // 0..15999
    const int rowBase = eTile * 16;
    const int half = lane >> 4;
    const int nlo = lane & 15;

    v8f zero = {};
    v8f acc[8];
#pragma unroll
    for (int ct = 0; ct < 8; ++ct) acc[ct] = zero;

#pragma unroll
    for (int c = 0; c < 4; ++c) {
        v16bf a = load_frag_bf16(ea16, DIM, rowBase, c * 32, lane);
#pragma unroll
        for (int ct = 0; ct < 8; ++ct) {
            v16bf b;
            const __bf16* rp = sW + (size_t)(ct * 16 + nlo) * DIM + c * 32;
#pragma unroll
            for (int i = 0; i < 8; ++i) {
                int k = frag_k(i, half);
                b[2 * i]     = rp[k];
                b[2 * i + 1] = rp[k + 1];
            }
            acc[ct] = __builtin_amdgcn_wmma_f32_16x16x32_bf16(false, a, false, b,
                                                              (short)0, acc[ct],
                                                              false, false);
        }
    }

#pragma unroll
    for (int r = 0; r < 8; ++r) {
        int e = rowBase + r + 8 * half;
        int s = ei[e], d = ei[NEDGES + e];
        int tgt = dir ? s : d;  // conv-dst (aggregation target)
        int oth = dir ? d : s;  // conv-src
        const float* Arow = A + (size_t)tgt * DIM;
        const float* Crow = C + (size_t)oth * DIM;
        float* Srow = S + (size_t)tgt * DIM;
#pragma unroll
        for (int ct = 0; ct < 8; ++ct) {
            int n = ct * 16 + nlo;
            float v = acc[ct][r] + Arow[n] + Crow[n] + sB[n];
            atomicAdd(&Srow[n], gelu_exact(v));
        }
    }
}

// ---------------------------------------------------------------------------
// Merge: y = x + GELU([x ; sum_fw/cnt ; sum_bw/cnt] @ Wmerge^T)   (pre-LN)
// ---------------------------------------------------------------------------
__global__ __launch_bounds__(256) void merge_kernel(
    const float* __restrict__ x,
    const float* __restrict__ sum_fw, const float* __restrict__ cnt_fw,
    const float* __restrict__ sum_bw, const float* __restrict__ cnt_bw,
    const float* __restrict__ Wmerge,  // [128, 384]
    float* __restrict__ y) {
    int wave = (blockIdx.x << 3) | (threadIdx.x >> 5);
    int lane = threadIdx.x & 31;
    int colTile = wave & 7;
    int nodeTile = wave >> 3;
    if (nodeTile >= NNODES / 16) return;

    int rowBase = nodeTile * 16, colBase = colTile * 16;
    int half = lane >> 4;
    int aRow = rowBase + (lane & 15);
    float invf = 1.0f / fmaxf(cnt_fw[aRow], 1.0f);
    float invb = 1.0f / fmaxf(cnt_bw[aRow], 1.0f);

    v8f acc = {};
#pragma unroll
    for (int c = 0; c < 12; ++c) {
        const float* src;
        float scale;
        int koff;
        if (c < 4)      { src = x;      scale = 1.0f; koff = c * 32; }
        else if (c < 8) { src = sum_fw; scale = invf; koff = c * 32 - DIM; }
        else            { src = sum_bw; scale = invb; koff = c * 32 - 2 * DIM; }
        const float* rp = src + (size_t)aRow * DIM + koff;
        v16bf a;
#pragma unroll
        for (int i = 0; i < 8; ++i) {
            int k = frag_k(i, half);
            a[2 * i]     = (__bf16)(rp[k] * scale);
            a[2 * i + 1] = (__bf16)(rp[k + 1] * scale);
        }
        v16bf b = load_frag(Wmerge, 3 * DIM, colBase, c * 32, lane);
        acc = __builtin_amdgcn_wmma_f32_16x16x32_bf16(false, a, false, b,
                                                      (short)0, acc, false, false);
    }
    int n = colBase + (lane & 15);
#pragma unroll
    for (int r = 0; r < 8; ++r) {
        size_t idx = (size_t)(rowBase + r + 8 * half) * DIM + n;
        y[idx] = x[idx] + gelu_exact(acc[r]);
    }
}

// ---------------------------------------------------------------------------
// LayerNorm: one wave per node row; 32 lanes x float4; shfl_xor reduction.
// ---------------------------------------------------------------------------
__global__ __launch_bounds__(256) void layernorm_kernel(
    const float* __restrict__ y, const float* __restrict__ g,
    const float* __restrict__ b, float* __restrict__ out) {
    int wave = (blockIdx.x << 3) | (threadIdx.x >> 5);
    int lane = threadIdx.x & 31;
    if (wave >= NNODES) return;
    int c = lane * 4;
    const float4 v = *reinterpret_cast<const float4*>(y + (size_t)wave * DIM + c);
    float s  = v.x + v.y + v.z + v.w;
    float sq = v.x * v.x + v.y * v.y + v.z * v.z + v.w * v.w;
#pragma unroll
    for (int off = 16; off >= 1; off >>= 1) {
        s  += __shfl_xor(s, off, 32);
        sq += __shfl_xor(sq, off, 32);
    }
    float mean = s * (1.0f / DIM);
    float var  = sq * (1.0f / DIM) - mean * mean;
    float rstd = rsqrtf(var + LN_EPS);
    const float4 go = *reinterpret_cast<const float4*>(g + c);
    const float4 bo = *reinterpret_cast<const float4*>(b + c);
    float4 o;
    o.x = (v.x - mean) * rstd * go.x + bo.x;
    o.y = (v.y - mean) * rstd * go.y + bo.y;
    o.z = (v.z - mean) * rstd * go.z + bo.z;
    o.w = (v.w - mean) * rstd * go.w + bo.w;
    *reinterpret_cast<float4*>(out + (size_t)wave * DIM + c) = o;
}

// ---------------------------------------------------------------------------
extern "C" void kernel_launch(void* const* d_in, const int* in_sizes, int n_in,
                              void* d_out, int out_size, void* d_ws, size_t ws_size,
                              hipStream_t stream) {
    const float* x0    = (const float*)d_in[0];
    const int*   ei    = (const int*)d_in[1];
    const float* eattr = (const float*)d_in[2];

    // Param leaves flattened in jax-pytree order (dict keys sorted):
    // per layer L, base = 3 + 15*L:
    //  +0 bw.e.W +1 bw.e.b +2 bw.m.W +3 bw.m.b +4 bw.n.W +5 bw.n.b
    //  +6 fw.e.W +7 fw.e.b +8 fw.m.W +9 fw.m.b +10 fw.n.W +11 fw.n.b
    //  +12 ln_b  +13 ln_g  +14 merge_W
    auto P = [&](int i) { return (const float*)d_in[i]; };

    // ---- workspace carve-up ----
    float* ws = (float*)d_ws;
    size_t off = 0;
    __bf16* W2b_fw = (__bf16*)(ws + off); off += DIM * DIM / 2;  // bf16 folded W2'
    __bf16* W2b_bw = (__bf16*)(ws + off); off += DIM * DIM / 2;
    float* W3_fw = ws + off; off += DIM * DIM;
    float* W3_bw = ws + off; off += DIM * DIM;
    float* b_fw  = ws + off; off += DIM;
    float* b_bw  = ws + off; off += DIM;
    float* cnt_fw = ws + off; off += NNODES;   // cnt_fw/cnt_bw contiguous
    float* cnt_bw = ws + off; off += NNODES;
    float* A_fw = ws + off; off += (size_t)NNODES * DIM;
    float* C_fw = ws + off; off += (size_t)NNODES * DIM;
    float* A_bw = ws + off; off += (size_t)NNODES * DIM;
    float* C_bw = ws + off; off += (size_t)NNODES * DIM;
    float* sum_fw = ws + off; off += (size_t)NNODES * DIM;  // sums contiguous
    float* sum_bw = ws + off; off += (size_t)NNODES * DIM;
    float* x_cur = ws + off; off += (size_t)NNODES * DIM;
    float* y_buf = ws + off; off += (size_t)NNODES * DIM;
    __bf16* ea16 = (__bf16*)(ws + off); off += (size_t)NEDGES * DIM / 2;

    // once-per-call prep: bf16 edge_attr + degree counts
    cvt_bf16_kernel<<<(NEDGES * DIM + 255) / 256, 256, 0, stream>>>(
        eattr, ea16, NEDGES * DIM);
    zero_kernel<<<(2 * NNODES + 255) / 256, 256, 0, stream>>>(cnt_fw, 2 * NNODES);
    count_kernel<<<(NEDGES + 255) / 256, 256, 0, stream>>>(ei, cnt_fw, cnt_bw);

    const float* xin = x0;
    for (int L = 0; L < NLAYERS; ++L) {
        const int base = 3 + 15 * L;
        const float *bweW = P(base + 0), *bweB = P(base + 1), *bwmW = P(base + 2),
                    *bwmB = P(base + 3), *bwnW = P(base + 4), *bwnB = P(base + 5);
        const float *fweW = P(base + 6), *fweB = P(base + 7), *fwmW = P(base + 8),
                    *fwmB = P(base + 9), *fwnW = P(base + 10), *fwnB = P(base + 11);
        const float *lnB = P(base + 12), *lnG = P(base + 13), *mergeW = P(base + 14);

        fold_kernel<<<128, 128, 0, stream>>>(fwmW, fwmB, fweW, fweB, fwnW, fwnB,
                                             W2b_fw, W3_fw, b_fw);
        fold_kernel<<<128, 128, 0, stream>>>(bwmW, bwmB, bweW, bweB, bwnW, bwnB,
                                             W2b_bw, W3_bw, b_bw);
        zero_kernel<<<(2 * NNODES * DIM + 255) / 256, 256, 0, stream>>>(
            sum_fw, 2 * NNODES * DIM);
        node_gemm_kernel<<<(NNODES / 16) * 8 * 4 / 8, 256, 0, stream>>>(
            xin, fwmW, W3_fw, bwmW, W3_bw, A_fw, C_fw, A_bw, C_bw);
        edge_msg_kernel<<<(NEDGES / 16 / 8) * 2, 256, 0, stream>>>(
            ea16, ei, W2b_fw, W2b_bw, b_fw, b_bw, A_fw, C_fw, A_bw, C_bw,
            sum_fw, sum_bw);
        merge_kernel<<<(NNODES / 16) * 8 / 8, 256, 0, stream>>>(
            xin, sum_fw, cnt_fw, sum_bw, cnt_bw, mergeW, y_buf);
        float* xout = (L == NLAYERS - 1) ? (float*)d_out : x_cur;
        layernorm_kernel<<<NNODES / 8, 256, 0, stream>>>(y_buf, lnG, lnB, xout);
        xin = xout;
    }
    (void)in_sizes; (void)n_in; (void)out_size; (void)ws_size;
}
